// DruseScorePKiV2_70540542869989
// MI455X (gfx1250) — compile-verified, matrix operand order checked by reading, not run on hardware
//
#include <hip/hip_runtime.h>
#include <hip/hip_bf16.h>

typedef __attribute__((ext_vector_type(16))) _Float16     v16h;
typedef __attribute__((ext_vector_type(8)))  _Float16     v8h;
typedef __attribute__((ext_vector_type(8)))  float        v8f;
typedef __attribute__((ext_vector_type(4)))  unsigned int v4u;
typedef __attribute__((ext_vector_type(8)))  int          v8i;
typedef __attribute__((ext_vector_type(4)))  int          v4i;

#define NB 8
#define NL 64
#define NP 512
#define NA 20
#define ND 128
#define NH 4
#define DH 32

#if defined(__has_builtin)
#if __has_builtin(__builtin_amdgcn_tensor_load_to_lds) && __has_builtin(__builtin_amdgcn_s_wait_tensorcnt)
#define USE_TDM 1
#endif
#endif
#ifndef USE_TDM
#define USE_TDM 0
#endif

// ---------------------------------------------------------------------------
// WMMA fragment load (16x16x32 f16). Operands stored row-major-by-K in LDS:
// tile[row][k]. Per-lane K pattern is two contiguous 8-half runs -> two
// aligned ds_load_b128 + shuffle.
// ---------------------------------------------------------------------------
__device__ __forceinline__ v16h frag_ld(const _Float16* tile, int ldt, int rbase, int lane) {
  int r  = rbase + (lane & 15);
  int kb = ((lane >> 4) & 1) << 3;
  const _Float16* p = tile + r * ldt + kb;
  v8h lo = *(const v8h*)(p);
  v8h hi = *(const v8h*)(p + 16);
  return __builtin_shufflevector(lo, hi, 0, 1, 2, 3, 4, 5, 6, 7, 8, 9, 10, 11, 12, 13, 14, 15);
}

// packed store of 4 halves (8 bytes) into LDS
__device__ __forceinline__ void st4h(_Float16* dst, float x, float y, float z, float w) {
  union { _Float16 h[4]; uint2 u; } pk;
  pk.h[0] = (_Float16)x; pk.h[1] = (_Float16)y;
  pk.h[2] = (_Float16)z; pk.h[3] = (_Float16)w;
  *(uint2*)dst = pk.u;
}

__device__ __forceinline__ float act_apply(float v, int act) {
  if (act == 1) return v / (1.f + __expf(-v));   // silu
  if (act == 2) return tanhf(v);                 // tanh
  return v;
}

// ---------------------------------------------------------------------------
// Generic batched WMMA GEMM: C = act(alpha * A*B(^T) + bias)
// Block tile 64x64, 256 threads = 8 waves, 2 output tiles per wave.
// LDS: At[m][k] 64x32, Bt[n][k] 64x32 (B always transposed into LDS).
// A and B staging fast-paths gated independently (fullM / fullN).
// ---------------------------------------------------------------------------
struct GemmP {
  const float* A; const float* Bm; const float* bias; float* C;
  int M, N, K;
  int lda, ldb, ldc;
  int sAb, sAh, sBb, sBh, sCb, sCh;
  int Hdiv;
  float alpha;
  int transB;
  int act;
};

__global__ __launch_bounds__(256) void gemm_wmma(GemmP g) {
  __shared__ __align__(16) _Float16 At[64 * 32];
  __shared__ __align__(16) _Float16 Bt[64 * 32];

  int z  = blockIdx.z;
  int zb = z / g.Hdiv, zh = z % g.Hdiv;
  const float* Ab = g.A  + zb * g.sAb + zh * g.sAh;
  const float* Bb = g.Bm + zb * g.sBb + zh * g.sBh;
  float*       Cb = g.C  + zb * g.sCb + zh * g.sCh;

  int m0 = blockIdx.y * 64, n0 = blockIdx.x * 64;
  int tid = threadIdx.x, lane = tid & 31, wave = tid >> 5;
  int t0  = wave * 2;
  int tm  = t0 >> 2;
  int tn0 = t0 & 3, tn1 = tn0 + 1;

  bool fullM = (m0 + 64 <= g.M);
  bool fullN = (n0 + 64 <= g.N);

  v8f acc0 = {}; v8f acc1 = {};

  for (int k0 = 0; k0 < g.K; k0 += 32) {
    bool fullK = (k0 + 32 <= g.K);
    if (fullM && fullK) {
      int r  = tid >> 3;          // 0..31 (x2)
      int c4 = (tid & 7) * 4;     // 0,4,..,28
      const float4 f0 = *(const float4*)(Ab + (m0 + r) * g.lda + k0 + c4);
      const float4 f1 = *(const float4*)(Ab + (m0 + r + 32) * g.lda + k0 + c4);
      st4h(&At[r * 32 + c4],        f0.x, f0.y, f0.z, f0.w);
      st4h(&At[(r + 32) * 32 + c4], f1.x, f1.y, f1.z, f1.w);
      if (k0 + 64 <= g.K)
        __builtin_prefetch((const void*)(Ab + (m0 + r) * g.lda + k0 + 32 + c4), 0, 1);
    } else {
      for (int idx = tid; idx < 2048; idx += 256) {
        int r = idx >> 5, k = idx & 31;
        int gm = m0 + r, gk = k0 + k;
        float v = (gm < g.M && gk < g.K) ? Ab[gm * g.lda + gk] : 0.f;
        At[r * 32 + k] = (_Float16)v;
      }
    }
    if (fullN && fullK) {
      if (g.transB) {
        int n  = tid >> 3;
        int c4 = (tid & 7) * 4;
        const float4 f0 = *(const float4*)(Bb + (n0 + n) * g.ldb + k0 + c4);
        const float4 f1 = *(const float4*)(Bb + (n0 + n + 32) * g.ldb + k0 + c4);
        st4h(&Bt[n * 32 + c4],        f0.x, f0.y, f0.z, f0.w);
        st4h(&Bt[(n + 32) * 32 + c4], f1.x, f1.y, f1.z, f1.w);
        if (k0 + 64 <= g.K)
          __builtin_prefetch((const void*)(Bb + (n0 + n) * g.ldb + k0 + 32 + c4), 0, 1);
      } else {
        // global rows along k, contiguous n -> transpose into LDS
        int k  = tid >> 4;          // 0..15 (x2)
        int n4 = (tid & 15) * 4;    // 0,4,..,60
        const float4 f0 = *(const float4*)(Bb + (k0 + k) * g.ldb + n0 + n4);
        const float4 f1 = *(const float4*)(Bb + (k0 + k + 16) * g.ldb + n0 + n4);
        Bt[(n4 + 0) * 32 + k] = (_Float16)f0.x;
        Bt[(n4 + 1) * 32 + k] = (_Float16)f0.y;
        Bt[(n4 + 2) * 32 + k] = (_Float16)f0.z;
        Bt[(n4 + 3) * 32 + k] = (_Float16)f0.w;
        Bt[(n4 + 0) * 32 + k + 16] = (_Float16)f1.x;
        Bt[(n4 + 1) * 32 + k + 16] = (_Float16)f1.y;
        Bt[(n4 + 2) * 32 + k + 16] = (_Float16)f1.z;
        Bt[(n4 + 3) * 32 + k + 16] = (_Float16)f1.w;
        if (k0 + 64 <= g.K)
          __builtin_prefetch((const void*)(Bb + (k0 + 32 + k) * g.ldb + n0 + n4), 0, 1);
      }
    } else {
      for (int idx = tid; idx < 2048; idx += 256) {
        int n = idx >> 5, k = idx & 31;
        int gn = n0 + n, gk = k0 + k;
        float v = 0.f;
        if (gn < g.N && gk < g.K)
          v = g.transB ? Bb[gn * g.ldb + gk] : Bb[gk * g.ldb + gn];
        Bt[n * 32 + k] = (_Float16)v;
      }
    }
    __syncthreads();

    v16h a  = frag_ld(At, 32, tm * 16, lane);
    v16h b0 = frag_ld(Bt, 32, tn0 * 16, lane);
    v16h b1 = frag_ld(Bt, 32, tn1 * 16, lane);
    acc0 = __builtin_amdgcn_wmma_f32_16x16x32_f16(false, a, false, b0, (short)0, acc0, false, false);
    acc1 = __builtin_amdgcn_wmma_f32_16x16x32_f16(false, a, false, b1, (short)0, acc1, false, false);
    __syncthreads();
  }

  int rowoff = ((lane >> 4) & 1) << 3;
#pragma unroll
  for (int t = 0; t < 2; ++t) {
    int tn = (t == 0) ? tn0 : tn1;
    int col = n0 + tn * 16 + (lane & 15);
    if (col >= g.N) continue;
#pragma unroll
    for (int r = 0; r < 8; ++r) {
      int row = m0 + tm * 16 + rowoff + r;
      if (row >= g.M) continue;
      float v = g.alpha * ((t == 0) ? acc0[r] : acc1[r]);
      if (g.bias) v += g.bias[col];
      v = act_apply(v, g.act);
      Cb[row * g.ldc + col] = v;
    }
  }
}

// ---------------------------------------------------------------------------
// Pairwise ligand-protein distances [B,L,P]
// ---------------------------------------------------------------------------
__global__ __launch_bounds__(256) void dist_kernel(const float* lig_pos, const float* prot_pos, float* dist) {
  int idx = blockIdx.x * 256 + threadIdx.x;
  if (idx >= NB * NL * NP) return;
  int p = idx & (NP - 1);
  int l = (idx >> 9) & (NL - 1);
  int b = idx >> 15;
  const float* lp = lig_pos  + (b * NL + l) * 3;
  const float* pp = prot_pos + (b * NP + p) * 3;
  float dx = lp[0] - pp[0], dy = lp[1] - pp[1], dz = lp[2] - pp[2];
  dist[idx] = sqrtf(dx * dx + dy * dy + dz * dz);
}

// ---------------------------------------------------------------------------
// RBF distance bias: bias[b,l,p,h] = sum_k exp(-10*(d-c_k)^2) * rbf_w[k,h] + rbf_b[h]
// ---------------------------------------------------------------------------
__global__ __launch_bounds__(256) void rbfbias_kernel(const float* dist, const float* rbf_w,
                                                      const float* rbf_b, float* bias) {
  int idx = blockIdx.x * 256 + threadIdx.x;
  if (idx >= NB * NL * NP) return;
  float d = dist[idx];
  float a0 = rbf_b[0], a1 = rbf_b[1], a2 = rbf_b[2], a3 = rbf_b[3];
#pragma unroll 5
  for (int k = 0; k < 50; ++k) {
    float c = (float)k * (10.f / 49.f);
    float t = d - c;
    float e = __expf(-10.f * t * t);
    a0 += e * rbf_w[k * 4 + 0];
    a1 += e * rbf_w[k * 4 + 1];
    a2 += e * rbf_w[k * 4 + 2];
    a3 += e * rbf_w[k * 4 + 3];
  }
  float4 o; o.x = a0; o.y = a1; o.z = a2; o.w = a3;
  *(float4*)&bias[idx * 4] = o;
}

// ---------------------------------------------------------------------------
// Masked softmax over P=512 for S[B,H,L,P]; one wave32 per row.
// ---------------------------------------------------------------------------
__global__ __launch_bounds__(256) void softmax_kernel(float* S, const float* bias, const int* prot_mask) {
  int row  = blockIdx.x * 8 + (threadIdx.x >> 5);
  int lane = threadIdx.x & 31;
  int l = row & (NL - 1);
  int h = (row >> 6) & (NH - 1);
  int b = row >> 8;
  float*       Sr = S + row * NP;
  const float* br = bias + (b * NL + l) * NP * 4 + h;
  const int*   mk = prot_mask + b * NP;

  float v[16];
  float mx = -3.0e38f;
#pragma unroll
  for (int i = 0; i < 16; ++i) {
    int p = lane + i * 32;
    float x = Sr[p] + br[p * 4];
    if (mk[p] == 0) x = -1e9f;
    v[i] = x;
    mx = fmaxf(mx, x);
  }
  for (int off = 16; off > 0; off >>= 1) mx = fmaxf(mx, __shfl_xor(mx, off, 32));
  float sm = 0.f;
#pragma unroll
  for (int i = 0; i < 16; ++i) {
    float e = __expf(v[i] - mx);
    v[i] = e;
    sm += e;
  }
  for (int off = 16; off > 0; off >>= 1) sm += __shfl_xor(sm, off, 32);
  float inv = 1.f / sm;
#pragma unroll
  for (int i = 0; i < 16; ++i) Sr[lane + i * 32] = v[i] * inv;
}

// ctx [B,H,L,32] -> [B*L, 128]
__global__ __launch_bounds__(256) void repack_ctx(const float* ctxBH, float* ctxL) {
  int idx = blockIdx.x * 256 + threadIdx.x;
  if (idx >= NB * NL * ND) return;
  int dcol = idx & (ND - 1);
  int row  = idx >> 7;
  int l = row & (NL - 1);
  int b = row >> 6;
  int h = dcol >> 5;
  int j = dcol & 31;
  ctxL[idx] = ctxBH[(((b * NH + h) * NL + l) * DH) + j];
}

// out = LayerNorm(x + y) * g + beta, rows of 128
__global__ __launch_bounds__(128) void ln_residual(const float* x, const float* y,
                                                   const float* g, const float* beta, float* out) {
  __shared__ float r1[128];
  __shared__ float r2[128];
  int row = blockIdx.x, d = threadIdx.x;
  float t = x[row * ND + d] + y[row * ND + d];
  r1[d] = t; r2[d] = t * t;
  __syncthreads();
  for (int off = 64; off > 0; off >>= 1) {
    if (d < off) { r1[d] += r1[d + off]; r2[d] += r2[d + off]; }
    __syncthreads();
  }
  float mu  = r1[0] * (1.f / ND);
  float var = r2[0] * (1.f / ND) - mu * mu;
  float rs  = rsqrtf(var + 1e-5f);
  out[row * ND + d] = (t - mu) * rs * g[d] + beta[d];
}

// attn_weights = mean over heads of last-layer softmaxed S
__global__ __launch_bounds__(256) void attn_avg(const float* S, float* out) {
  int idx = blockIdx.x * 256 + threadIdx.x;
  if (idx >= NB * NL * NP) return;
  int p = idx & (NP - 1);
  int l = (idx >> 9) & (NL - 1);
  int b = idx >> 15;
  float s = 0.f;
#pragma unroll
  for (int h = 0; h < NH; ++h) s += S[((b * NH + h) * NL + l) * NP + p];
  out[idx] = 0.25f * s;
}

// gated attention pooling -> complex_repr [B,128]
__global__ __launch_bounds__(128) void pool_kernel(const float* g1, const float* gw2, const float* gb2,
                                                   const int* lig_mask, const float* lig, float* cx) {
  __shared__ float s[64];
  __shared__ float tmp[64];
  int b = blockIdx.x, t = threadIdx.x;
  if (t < 64) {
    float a = gb2[0];
    const float* gr = g1 + (b * NL + t) * 64;
    for (int j = 0; j < 64; ++j) a += gr[j] * gw2[j];
    if (lig_mask[b * NL + t] == 0) a = -1e9f;
    s[t] = a;
  }
  __syncthreads();
  if (t < 64) tmp[t] = s[t];
  __syncthreads();
  for (int off = 32; off > 0; off >>= 1) {
    if (t < off) tmp[t] = fmaxf(tmp[t], tmp[t + off]);
    __syncthreads();
  }
  float mx = tmp[0];
  __syncthreads();
  if (t < 64) { s[t] = __expf(s[t] - mx); tmp[t] = s[t]; }
  __syncthreads();
  for (int off = 32; off > 0; off >>= 1) {
    if (t < off) tmp[t] += tmp[t + off];
    __syncthreads();
  }
  float inv = 1.f / tmp[0];
  __syncthreads();
  float acc = 0.f;
  for (int l = 0; l < NL; ++l) acc += s[l] * inv * lig[(b * NL + l) * ND + t];
  cx[b * ND + t] = acc;
}

// affinity / confidence / dock heads -> out[0..23]
__global__ __launch_bounds__(128) void heads_kernel(const float* cx,
    const float* aw1, const float* ab1, const float* aw2, const float* ab2,
    const float* cw1, const float* cb1, const float* cw2, const float* cb2, float* out) {
  __shared__ float red[128];
  int b = blockIdx.x, t = threadIdx.x;
  const float* c = cx + b * ND;

  float a = ab1[t];
  for (int k = 0; k < ND; ++k) a += c[k] * aw1[k * ND + t];
  a = a / (1.f + __expf(-a));
  red[t] = a * aw2[t];
  __syncthreads();
  for (int off = 64; off > 0; off >>= 1) {
    if (t < off) red[t] += red[t + off];
    __syncthreads();
  }
  float pkd = red[0] + ab2[0];
  __syncthreads();

  float h2 = 0.f;
  if (t < 64) {
    float a2 = cb1[t];
    for (int k = 0; k < ND; ++k) a2 += c[k] * cw1[k * 64 + t];
    a2 = a2 / (1.f + __expf(-a2));
    h2 = a2 * cw2[t];
  }
  red[t] = (t < 64) ? h2 : 0.f;
  __syncthreads();
  for (int off = 64; off > 0; off >>= 1) {
    if (t < off) red[t] += red[t + off];
    __syncthreads();
  }
  float conf = 1.f / (1.f + __expf(-(red[0] + cb2[0])));
  if (t == 0) {
    out[b]      = pkd;
    out[8 + b]  = conf;
    out[16 + b] = pkd * conf;
  }
}

// ---------------------------------------------------------------------------
// Fused interaction head. Block = (b, 8 ligand rows, 16 protein cols) = 128 pairs.
// w_r tile (50x128 f32) staged to LDS via Tensor Data Mover (overlapped with
// the RBF exp tile), then converted/transposed to f16. RT[pair][k] 128x64 f16,
// WRt[n][k] 128x64 f16; 16 WMMAs/wave; + row terms; silu -> f16 LDS (aliased);
// tiny [128,5] GEMM + sigmoid to d_out.
// ---------------------------------------------------------------------------
__global__ __launch_bounds__(256) void interaction_kernel(const float* dist, const float* Ilig,
                                                          const float* Iprot, const float* wr,
                                                          const float* w2, const float* b2, float* out) {
  __shared__ __align__(16) _Float16 smh[16384];   // 32KB, phase-aliased
#if USE_TDM
  __shared__ __align__(16) float WRf[50 * ND];    // 25.6KB TDM landing buffer
#endif
  _Float16* RT  = smh;          // [128][64]
  _Float16* WRt = smh + 8192;   // [128][64]  (n-major, k-minor)

  int b = blockIdx.z, l0 = blockIdx.y * 8, p0 = blockIdx.x * 16;
  int tid = threadIdx.x, lane = tid & 31, wave = tid >> 5;

#if USE_TDM
  // Issue the TDM load of w_r (50 rows x 128 f32) into LDS from wave 0,
  // overlapped with the RBF tile computation below.
  if (wave == 0) {
    unsigned lds_off = (unsigned)(size_t)(void*)WRf;
    unsigned long long ga = (unsigned long long)(size_t)wr;
    v4u g0 = { 1u,                                   // count=1 (valid descriptor)
               lds_off,                              // lds_addr
               (unsigned)(ga & 0xffffffffu),         // global_addr[31:0]
               (unsigned)((ga >> 32) & 0x1ffffffu)   // global_addr[56:32]
                 | 0x80000000u };                    // type=2 ("image")
    v8i g1 = { (int)(2u << 16),        // data_size=2 (4 bytes)
               (int)(128u << 16),      // tensor_dim0[15:0]=128 @ bits[63:48]
               (int)(50u << 16),       // tensor_dim1[15:0]=50  @ bits[95:80]
               (int)(128u << 16),      // tile_dim0=128 @ bits[127:112]
               (int)50,                // tile_dim1=50  @ bits[143:128]
               (int)128,               // tensor_dim0_stride[31:0]=128
               0, 0 };
    v4i g2 = { 0, 0, 0, 0 };
    v4i g3 = { 0, 0, 0, 0 };
    v8i g4 = { 0, 0, 0, 0, 0, 0, 0, 0 };
    __builtin_amdgcn_tensor_load_to_lds(g0, g1, g2, g3, g4, 0);
  }
#endif

  // RBF tile: thread = (pair, half); one distance load, 32 exps, packed stores
  {
    int pair = tid & 127, half = tid >> 7;
    int ll = pair >> 4, pp = pair & 15;
    float d = dist[(b * NL + l0 + ll) * NP + p0 + pp];
    int kbase = half * 32;
#pragma unroll
    for (int j = 0; j < 32; j += 4) {
      float e[4];
#pragma unroll
      for (int i = 0; i < 4; ++i) {
        int k = kbase + j + i;
        float c = (float)k * (10.f / 49.f);
        float t = d - c;
        e[i] = (k < 50) ? __expf(-10.f * t * t) : 0.f;
      }
      st4h(&RT[pair * 64 + kbase + j], e[0], e[1], e[2], e[3]);
    }
  }

#if USE_TDM
  if (wave == 0) __builtin_amdgcn_s_wait_tensorcnt(0);
  __syncthreads();
  // convert/transpose from LDS f32: WRf[k][n] -> WRt[n][k]
  for (int i = tid; i < 50 * 32; i += 256) {
    int k = i >> 5, n4 = (i & 31) * 4;
    const float4 f = *(const float4*)&WRf[k * ND + n4];
    WRt[(n4 + 0) * 64 + k] = (_Float16)f.x;
    WRt[(n4 + 1) * 64 + k] = (_Float16)f.y;
    WRt[(n4 + 2) * 64 + k] = (_Float16)f.z;
    WRt[(n4 + 3) * 64 + k] = (_Float16)f.w;
  }
#else
  // fallback: transpose straight from global
  for (int i = tid; i < 50 * 32; i += 256) {
    int k = i >> 5, n4 = (i & 31) * 4;
    const float4 f = *(const float4*)(wr + k * ND + n4);
    WRt[(n4 + 0) * 64 + k] = (_Float16)f.x;
    WRt[(n4 + 1) * 64 + k] = (_Float16)f.y;
    WRt[(n4 + 2) * 64 + k] = (_Float16)f.z;
    WRt[(n4 + 3) * 64 + k] = (_Float16)f.w;
  }
#endif
  for (int i = tid; i < 128 * 14; i += 256) {
    int n = i & 127, k = 50 + (i >> 7);
    WRt[n * 64 + k] = (_Float16)0.f;
  }
  __syncthreads();

  v8f zero = {};
  v8f acc[8];
#pragma unroll
  for (int i = 0; i < 8; ++i) acc[i] = zero;

#pragma unroll
  for (int ks = 0; ks < 64; ks += 32) {
    v16h a = frag_ld(RT + ks, 64, wave * 16, lane);
#pragma unroll
    for (int tn = 0; tn < 8; ++tn) {
      v16h bb = frag_ld(WRt + ks, 64, tn * 16, lane);
      acc[tn] = __builtin_amdgcn_wmma_f32_16x16x32_f16(false, a, false, bb, (short)0, acc[tn], false, false);
    }
  }
  __syncthreads();   // done reading RT/WRt

  _Float16* PH = smh; // [128][128] (aliases RT/WRt)
  int rowoff = ((lane >> 4) & 1) << 3;
#pragma unroll
  for (int tn = 0; tn < 8; ++tn) {
    int col = tn * 16 + (lane & 15);
#pragma unroll
    for (int r = 0; r < 8; ++r) {
      int row = wave * 16 + rowoff + r;
      int ll = row >> 4, pp = row & 15;
      float v = acc[tn][r]
              + Ilig[(b * NL + l0 + ll) * ND + col]
              + Iprot[(b * NP + p0 + pp) * ND + col];
      v = v / (1.f + __expf(-v));
      PH[row * ND + col] = (_Float16)v;
    }
  }
  __syncthreads();

  for (int t = tid; t < 128 * 5; t += 256) {
    int pair = t / 5, o = t % 5;
    float a2 = b2[o];
    const _Float16* ph = PH + pair * ND;
    for (int k = 0; k < ND; ++k) a2 += (float)ph[k] * w2[k * 5 + o];
    float sg = 1.f / (1.f + __expf(-a2));
    int ll = pair >> 4, pp = pair & 15;
    out[((b * NL + l0 + ll) * NP + p0 + pp) * 5 + o] = sg;
  }
}

// ---------------------------------------------------------------------------
// Host orchestration
// ---------------------------------------------------------------------------
static void gemm(hipStream_t s, const float* A, const float* Bm, const float* bias, float* C,
                 int M, int N, int K, int lda, int ldb, int ldc,
                 int sAb, int sAh, int sBb, int sBh, int sCb, int sCh,
                 int Z, int Hdiv, float alpha, int transB, int act) {
  GemmP g;
  g.A = A; g.Bm = Bm; g.bias = bias; g.C = C;
  g.M = M; g.N = N; g.K = K; g.lda = lda; g.ldb = ldb; g.ldc = ldc;
  g.sAb = sAb; g.sAh = sAh; g.sBb = sBb; g.sBh = sBh; g.sCb = sCb; g.sCh = sCh;
  g.Hdiv = Hdiv; g.alpha = alpha; g.transB = transB; g.act = act;
  dim3 grid((N + 63) / 64, (M + 63) / 64, Z);
  gemm_wmma<<<grid, 256, 0, s>>>(g);
}

extern "C" void kernel_launch(void* const* d_in, const int* in_sizes, int n_in,
                              void* d_out, int out_size, void* d_ws, size_t ws_size,
                              hipStream_t stream) {
  (void)in_sizes; (void)n_in; (void)out_size; (void)ws_size;

  const float* prot_x   = (const float*)d_in[0];
  const float* lig_x    = (const float*)d_in[1];
  const float* prot_pos = (const float*)d_in[2];
  const float* lig_pos  = (const float*)d_in[3];
  const int*   prot_mask= (const int*)d_in[4];
  const int*   lig_mask = (const int*)d_in[5];
  auto W = [&](int i) { return (const float*)d_in[i]; };
  // params (insertion order): 6..9 prot mlp, 10..13 lig mlp,
  // 14..25 layer0 {q_w,q_b,k_w,k_b,v_w,v_b,rbf_w,rbf_b,o_w,o_b,ln_g,ln_b}, 26..37 layer1,
  // 38..41 gate, 42..45 aff, 46..49 conf, 50..53 int.

  float* out = (float*)d_out;
  float* ws  = (float*)d_ws;
  size_t off = 0;
  auto alloc = [&](size_t n) { float* p = ws + off; off += n; return p; };

  const int BLP = NB * NL * NP;       // 262144
  float* dist   = alloc(BLP);
  float* rbfb   = alloc((size_t)BLP * 4);
  float* prot_h = alloc((size_t)NB * NP * ND);
  float* tmpP   = alloc((size_t)NB * NP * ND);
  float* lig_h  = alloc((size_t)NB * NL * ND);
  float* tmpL   = alloc((size_t)NB * NL * ND);
  float* ligA   = alloc((size_t)NB * NL * ND);
  float* ligB   = alloc((size_t)NB * NL * ND);
  float* Qp     = alloc((size_t)NB * NL * ND);
  float* Kp     = alloc((size_t)NB * NP * ND);
  float* Vp     = alloc((size_t)NB * NP * ND);
  float* S      = alloc((size_t)NB * NH * NL * NP); // 1048576
  float* ctxBH  = alloc((size_t)NB * NH * NL * DH);
  float* ctxL   = alloc((size_t)NB * NL * ND);
  float* ctxo   = alloc((size_t)NB * NL * ND);
  float* g1     = alloc((size_t)NB * NL * 64);
  float* cx     = alloc((size_t)NB * ND);
  float* Ilig   = alloc((size_t)NB * NL * ND);
  float* Iprot  = alloc((size_t)NB * NP * ND);

  const int MP = NB * NP;   // 4096 protein rows
  const int ML = NB * NL;   // 512 ligand rows

  // distances (needed before softmax / interaction)
  dist_kernel<<<(BLP + 255) / 256, 256, 0, stream>>>(lig_pos, prot_pos, dist);

  // embeddings
  gemm(stream, prot_x, W(6), W(7), tmpP,  MP, ND, NA, NA, ND, ND, 0,0,0,0,0,0, 1,1, 1.f, 0, 1);
  gemm(stream, tmpP,  W(8), W(9), prot_h, MP, ND, ND, ND, ND, ND, 0,0,0,0,0,0, 1,1, 1.f, 0, 0);
  gemm(stream, lig_x, W(10), W(11), tmpL, ML, ND, NA, NA, ND, ND, 0,0,0,0,0,0, 1,1, 1.f, 0, 1);
  gemm(stream, tmpL,  W(12), W(13), lig_h,ML, ND, ND, ND, ND, ND, 0,0,0,0,0,0, 1,1, 1.f, 0, 0);

  const float* cur = lig_h;
  float* nxt[2] = { ligA, ligB };
  for (int li = 0; li < 2; ++li) {
    int pi = 14 + li * 12;
    rbfbias_kernel<<<(BLP + 255) / 256, 256, 0, stream>>>(dist, W(pi + 6), W(pi + 7), rbfb);

    gemm(stream, cur,    W(pi + 0), W(pi + 1), Qp, ML, ND, ND, ND, ND, ND, 0,0,0,0,0,0, 1,1, 1.f, 0, 0);
    gemm(stream, prot_h, W(pi + 2), W(pi + 3), Kp, MP, ND, ND, ND, ND, ND, 0,0,0,0,0,0, 1,1, 1.f, 0, 0);
    gemm(stream, prot_h, W(pi + 4), W(pi + 5), Vp, MP, ND, ND, ND, ND, ND, 0,0,0,0,0,0, 1,1, 1.f, 0, 0);

    // scores: per (b,h): Q_h[64,32] x K_h^T -> S[b,h,64,512], alpha = 1/sqrt(32)
    gemm(stream, Qp, Kp, nullptr, S, NL, NP, DH, ND, ND, NP,
         NL * ND, DH, NP * ND, DH, NH * NL * NP, NL * NP,
         NB * NH, NH, 0.17677669529663687f, 1, 0);

    softmax_kernel<<<(NB * NH * NL) / 8, 256, 0, stream>>>(S, rbfb, prot_mask);

    // ctx: per (b,h): S[64,512] x V_h[512,32] -> ctxBH[b,h,64,32]
    gemm(stream, S, Vp, nullptr, ctxBH, NL, DH, NP, NP, ND, DH,
         NH * NL * NP, NL * NP, NP * ND, DH, NH * NL * DH, NL * DH,
         NB * NH, NH, 1.f, 0, 0);

    repack_ctx<<<(ML * ND + 255) / 256, 256, 0, stream>>>(ctxBH, ctxL);
    gemm(stream, ctxL, W(pi + 8), W(pi + 9), ctxo, ML, ND, ND, ND, ND, ND, 0,0,0,0,0,0, 1,1, 1.f, 0, 0);
    ln_residual<<<ML, 128, 0, stream>>>(cur, ctxo, W(pi + 10), W(pi + 11), nxt[li]);
    cur = nxt[li];
  }

  // attn_weights = mean over heads of last-layer S
  attn_avg<<<(BLP + 255) / 256, 256, 0, stream>>>(S, out + 24 + (long)BLP * 5);

  // gated pooling
  gemm(stream, cur, W(38), W(39), g1, ML, 64, ND, ND, 64, 64, 0,0,0,0,0,0, 1,1, 1.f, 0, 2);
  pool_kernel<<<NB, 128, 0, stream>>>(g1, W(40), W(41), lig_mask, cur, cx);

  // pkd / conf / dock
  heads_kernel<<<NB, 128, 0, stream>>>(cx, W(42), W(43), W(44), W(45),
                                       W(46), W(47), W(48), W(49), out);

  // interaction head terms
  const float* int_w1 = W(50);
  gemm(stream, cur,    int_w1,            W(51),   Ilig,  ML, ND, ND, ND, ND, ND, 0,0,0,0,0,0, 1,1, 1.f, 0, 0);
  gemm(stream, prot_h, int_w1 + ND * ND,  nullptr, Iprot, MP, ND, ND, ND, ND, ND, 0,0,0,0,0,0, 1,1, 1.f, 0, 0);

  dim3 igrid(NP / 16, NL / 8, NB);
  interaction_kernel<<<igrid, 256, 0, stream>>>(dist, Ilig, Iprot,
                                                int_w1 + 2 * ND * ND, W(52), W(53), out + 24);
}
